// BiMamba4KT_89129161326594
// MI455X (gfx1250) — matrix operand order repaired and checked by reading, not compiled, not run on hardware
//
#include <hip/hip_runtime.h>
#include <hip/hip_bf16.h>
#include <math.h>

// ---- model constants ----
#define QUES  3162
#define EMB   256
#define DST   16
#define DIN   512
#define DTR   16
#define DCONV 4
#define BB    32
#define SS    512
#define TTOK  (BB*SS)          // 16384 tokens
#define QPAD  3168             // QUES padded to multiple of 16

typedef __attribute__((ext_vector_type(16))) _Float16 v16h;
typedef __attribute__((ext_vector_type(8)))  _Float16 h8;
typedef __attribute__((ext_vector_type(8)))  float    v8f;

#define GFLAG_ACC  1
#define GFLAG_GELU 2
#define GFLAG_SP   4

__device__ __forceinline__ h8 ld8(const _Float16* p) { return *(const h8*)p; }

__device__ __forceinline__ v16h packfrag(h8 lo, h8 hi2) {
    v16h r;
    #pragma unroll
    for (int i = 0; i < 8; i++) { r[i] = lo[i]; r[i + 8] = hi2[i]; }
    return r;
}

#define WMMA(AF, BF, ACC) __builtin_amdgcn_wmma_f32_16x16x32_f16( \
    /*neg_a=*/false, (AF), /*neg_b=*/false, (BF), /*c_mod=*/(short)0, (ACC), \
    /*reuse_a=*/false, /*reuse_b=*/false)

// -------- weight transpose + f32->f16 convert (with zero padding) --------
// in: K x N (row major f32)   out: Npad x Kpad (row major f16), B^T layout
__global__ void k_transpose_f16(const float* __restrict__ in, _Float16* __restrict__ out,
                                int K, int N, int Kpad, int Npad) {
    int idx = blockIdx.x * blockDim.x + threadIdx.x;
    if (idx >= Npad * Kpad) return;
    int n = idx / Kpad, k = idx % Kpad;
    float v = (n < N && k < K) ? in[(size_t)k * N + n] : 0.0f;
    out[idx] = (_Float16)v;
}

// -------- embedding gather + LayerNorm (one wave32 per token, EMB=256) --------
__global__ void k_embed_ln(const int* __restrict__ qa, const float* __restrict__ tab,
                           const float* __restrict__ g, const float* __restrict__ b,
                           float* __restrict__ out, int T, float eps) {
    int wave = (blockIdx.x * blockDim.x + threadIdx.x) >> 5;
    int lane = threadIdx.x & 31;
    if (wave >= T) return;
    const float* row = tab + (size_t)qa[wave] * EMB;
    float vals[8]; float s = 0.f;
    #pragma unroll
    for (int i = 0; i < 8; i++) { vals[i] = row[lane + 32*i]; s += vals[i]; }
    #pragma unroll
    for (int m = 16; m > 0; m >>= 1) s += __shfl_xor(s, m, 32);
    float mean = s * (1.0f / EMB);
    float v = 0.f;
    #pragma unroll
    for (int i = 0; i < 8; i++) { float d = vals[i] - mean; v += d * d; }
    #pragma unroll
    for (int m = 16; m > 0; m >>= 1) v += __shfl_xor(v, m, 32);
    float r = rsqrtf(v * (1.0f / EMB) + eps);
    float* orow = out + (size_t)wave * EMB;
    #pragma unroll
    for (int i = 0; i < 8; i++) {
        int c = lane + 32*i;
        orow[c] = (vals[i] - mean) * r * g[c] + b[c];
    }
}

// -------- generic LayerNorm over EMB=256 (optional residual, f32/f16 outs) --------
__global__ void k_ln(const float* __restrict__ x, const float* __restrict__ res,
                     const float* __restrict__ g, const float* __restrict__ b,
                     float* __restrict__ outf, _Float16* __restrict__ outh,
                     int T, float eps) {
    int wave = (blockIdx.x * blockDim.x + threadIdx.x) >> 5;
    int lane = threadIdx.x & 31;
    if (wave >= T) return;
    const float* row = x + (size_t)wave * EMB;
    const float* rrow = res ? (res + (size_t)wave * EMB) : nullptr;
    float vals[8]; float s = 0.f;
    #pragma unroll
    for (int i = 0; i < 8; i++) {
        int c = lane + 32*i;
        vals[i] = row[c] + (rrow ? rrow[c] : 0.f);
        s += vals[i];
    }
    #pragma unroll
    for (int m = 16; m > 0; m >>= 1) s += __shfl_xor(s, m, 32);
    float mean = s * (1.0f / EMB);
    float v = 0.f;
    #pragma unroll
    for (int i = 0; i < 8; i++) { float d = vals[i] - mean; v += d * d; }
    #pragma unroll
    for (int m = 16; m > 0; m >>= 1) v += __shfl_xor(v, m, 32);
    float r = rsqrtf(v * (1.0f / EMB) + eps);
    #pragma unroll
    for (int i = 0; i < 8; i++) {
        int c = lane + 32*i;
        float o = (vals[i] - mean) * r * g[c] + b[c];
        if (outf) outf[(size_t)wave * EMB + c] = o;
        if (outh) outh[(size_t)wave * EMB + c] = (_Float16)o;
    }
}

// -------- GEMM epilogue for one 16x16 tile's per-lane 8 results --------
__device__ __forceinline__ void gemm_epi(v8f acc, int mb, int n, int hi,
                                         const float* __restrict__ bias,
                                         float* __restrict__ Cf, _Float16* __restrict__ Ch,
                                         int Nreal, int flags) {
    if (n >= Nreal) return;
    float bv = bias ? bias[n] : 0.f;
    #pragma unroll
    for (int r = 0; r < 8; r++) {
        int m = mb + r + 8 * hi;  // C layout: VGPR r -> row r (lanes<16) / 8+r (lanes>=16)
        size_t idx = (size_t)m * Nreal + n;
        float v = acc[r] + bv;
        if (flags & GFLAG_ACC)  v += Cf[idx];
        if (flags & GFLAG_GELU) v = 0.5f * v * (1.0f + erff(v * 0.70710678118654752f));
        if (flags & GFLAG_SP)   v = (v > 20.f) ? v : log1pf(expf(v));
        if (Cf) Cf[idx] = v;
        if (Ch) Ch[idx] = (_Float16)v;
    }
}

// -------- WMMA GEMM: C[M x Nreal] = A[M x K] * BT[Npad x K]^T (+bias, epilogue) ----
// One wave computes a 32x32 C block (2x2 tiles, 4 WMMAs/K-step) with register
// double-buffering so next K-step loads overlap the current matrix ops.
// 4 waves/block cover 32 rows x 128 cols.
__global__ void k_gemm_wmma(const _Float16* __restrict__ A, const _Float16* __restrict__ BT,
                            const float* __restrict__ bias,
                            float* __restrict__ Cf, _Float16* __restrict__ Ch,
                            int M, int Npad, int K, int Nreal, int flags) {
    int wid  = threadIdx.x >> 5;
    int lane = threadIdx.x & 31;
    int mbase = blockIdx.y * 32;
    int nbase = (blockIdx.x * 4 + wid) * 32;
    if (nbase >= Npad) return;

    int mr = lane & 15;
    int hi = lane >> 4;  // 0 or 1
    const _Float16* arow0 = A + (size_t)(mbase + mr) * K;
    const _Float16* arow1 = arow0 + (size_t)16 * K;
    int n0 = nbase + mr;
    int n1 = nbase + 16 + mr;
    int n1c = (n1 < Npad) ? n1 : (Npad - 1);   // clamp loads for ragged N
    const _Float16* brow0 = BT + (size_t)n0  * K;
    const _Float16* brow1 = BT + (size_t)n1c * K;

    // A 16x32 fragment: lanes 0-15 hold K {0..7,16..23}; lanes 16-31 hold {8..15,24..31}
    const int aoff0 = hi * 8, aoff1 = 16 + hi * 8;
    // B 32x16 fragment: lane n<16 -> column n, K {0..7,16..23}; lane n+16 -> K {8..15,24..31}
    const int boff0 = hi * 8, boff1 = 16 + hi * 8;

    h8 a0l = ld8(arow0 + aoff0), a0h = ld8(arow0 + aoff1);
    h8 a1l = ld8(arow1 + aoff0), a1h = ld8(arow1 + aoff1);
    h8 b0l = ld8(brow0 + boff0), b0h = ld8(brow0 + boff1);
    h8 b1l = ld8(brow1 + boff0), b1h = ld8(brow1 + boff1);

    v8f acc00 = {}, acc01 = {}, acc10 = {}, acc11 = {};
    for (int kb = 32; kb < K; kb += 32) {
        // issue next K-step loads before current WMMAs (register double buffer)
        h8 na0l = ld8(arow0 + kb + aoff0), na0h = ld8(arow0 + kb + aoff1);
        h8 na1l = ld8(arow1 + kb + aoff0), na1h = ld8(arow1 + kb + aoff1);
        h8 nb0l = ld8(brow0 + kb + boff0), nb0h = ld8(brow0 + kb + boff1);
        h8 nb1l = ld8(brow1 + kb + boff0), nb1h = ld8(brow1 + kb + boff1);
        __builtin_prefetch(arow0 + kb + 64, 0, 2);
        __builtin_prefetch(brow0 + kb + 64, 0, 2);
        v16h af0 = packfrag(a0l, a0h), af1 = packfrag(a1l, a1h);
        v16h bf0 = packfrag(b0l, b0h), bf1 = packfrag(b1l, b1h);
        acc00 = WMMA(af0, bf0, acc00);
        acc01 = WMMA(af0, bf1, acc01);
        acc10 = WMMA(af1, bf0, acc10);
        acc11 = WMMA(af1, bf1, acc11);
        a0l = na0l; a0h = na0h; a1l = na1l; a1h = na1h;
        b0l = nb0l; b0h = nb0h; b1l = nb1l; b1h = nb1h;
    }
    {
        v16h af0 = packfrag(a0l, a0h), af1 = packfrag(a1l, a1h);
        v16h bf0 = packfrag(b0l, b0h), bf1 = packfrag(b1l, b1h);
        acc00 = WMMA(af0, bf0, acc00);
        acc01 = WMMA(af0, bf1, acc01);
        acc10 = WMMA(af1, bf0, acc10);
        acc11 = WMMA(af1, bf1, acc11);
    }

    gemm_epi(acc00, mbase,      n0, hi, bias, Cf, Ch, Nreal, flags);
    gemm_epi(acc01, mbase,      n1, hi, bias, Cf, Ch, Nreal, flags);
    gemm_epi(acc10, mbase + 16, n0, hi, bias, Cf, Ch, Nreal, flags);
    gemm_epi(acc11, mbase + 16, n1, hi, bias, Cf, Ch, Nreal, flags);
}

// -------- depthwise causal conv (dir 0: taps t-3..t; dir 1: taps t..t+3 reversed) + SiLU
__global__ void k_conv_silu(const float* __restrict__ xz, const float* __restrict__ cw,
                            const float* __restrict__ cb,
                            float* __restrict__ xs_f, _Float16* __restrict__ xs_h, int dir) {
    int idx = blockIdx.x * blockDim.x + threadIdx.x;
    if (idx >= BB * SS * DIN) return;
    int c = idx & (DIN - 1);
    int t = (idx >> 9) & (SS - 1);
    int b = idx >> 18;
    float s = cb[c];
    #pragma unroll
    for (int k = 0; k < DCONV; k++) {
        int tt = dir ? (t + 3 - k) : (t - 3 + k);
        if (tt >= 0 && tt < SS)
            s += cw[c * DCONV + k] * xz[((size_t)b * SS + tt) * (2 * DIN) + c];
    }
    float v = s * (1.0f / (1.0f + expf(-s)));  // silu
    size_t o = ((size_t)b * SS + t) * DIN + c;
    xs_f[o] = v;
    xs_h[o] = (_Float16)v;
}

// -------- pack dt-input: (T,48) cols 0..15 -> (T,32) f16, zero padded --------
__global__ void k_dtin(const float* __restrict__ dbc, _Float16* __restrict__ dtin, int T) {
    int idx = blockIdx.x * blockDim.x + threadIdx.x;
    if (idx >= T * 32) return;
    int t = idx >> 5, j = idx & 31;
    dtin[idx] = (_Float16)((j < DTR) ? dbc[(size_t)t * 48 + j] : 0.f);
}

// -------- selective scan: one thread per (batch, channel); B/C staged in LDS ----
__global__ void k_scan(const float* __restrict__ dt, const float* __restrict__ xs_f,
                       const float* __restrict__ dbc, const float* __restrict__ A_log,
                       const float* __restrict__ Dp, const float* __restrict__ xz,
                       _Float16* __restrict__ y_h, int dir) {
    int b = blockIdx.x >> 1;                         // 64 blocks = 32 batches x 2 chunks
    int c = ((blockIdx.x & 1) << 8) + threadIdx.x;   // 256 threads -> 512 channels
    float Av[DST], hst[DST];
    #pragma unroll
    for (int s = 0; s < DST; s++) { Av[s] = -expf(A_log[c * DST + s]); hst[s] = 0.f; }
    float Dc = Dp[c];
    __shared__ float sB[DST], sC[DST];
    for (int step = 0; step < SS; step++) {
        int t = dir ? (SS - 1 - step) : step;
        size_t tok = (size_t)b * SS + t;
        __syncthreads();
        if (threadIdx.x < 32) {
            float v = dbc[tok * 48 + DTR + threadIdx.x];
            if (threadIdx.x < DST) sB[threadIdx.x] = v; else sC[threadIdx.x - DST] = v;
        }
        __syncthreads();
        float dtv = dt[tok * DIN + c];
        float xv  = xs_f[tok * DIN + c];
        float dtx = dtv * xv;
        float y = 0.f;
        #pragma unroll
        for (int s = 0; s < DST; s++) {
            hst[s] = expf(dtv * Av[s]) * hst[s] + dtx * sB[s];
            y += hst[s] * sC[s];
        }
        y += xv * Dc;
        float z = xz[tok * (2 * DIN) + DIN + c];
        float sil = z * (1.0f / (1.0f + expf(-z)));
        y_h[tok * DIN + c] = (_Float16)(y * sil);
    }
}

// ------------------------------- launcher -------------------------------
extern "C" void kernel_launch(void* const* d_in, const int* in_sizes, int n_in,
                              void* d_out, int out_size, void* d_ws, size_t ws_size,
                              hipStream_t stream) {
    const int    T = TTOK;
    // inputs in setup_inputs() order
    const int*   qa      = (const int*)  d_in[0];
    const float* qa_tab  = (const float*)d_in[4];
    const float* ln0_g   = (const float*)d_in[5];
    const float* ln0_b   = (const float*)d_in[6];
    const float* in_w    = (const float*)d_in[7];
    const float* conv_w  = (const float*)d_in[8];
    const float* conv_b  = (const float*)d_in[9];
    const float* xp_w    = (const float*)d_in[10];
    const float* dt_w    = (const float*)d_in[11];
    const float* dt_b    = (const float*)d_in[12];
    const float* A_log   = (const float*)d_in[13];
    const float* Dp      = (const float*)d_in[14];
    const float* out_w   = (const float*)d_in[15];
    const float* n1_g    = (const float*)d_in[16];
    const float* n1_b    = (const float*)d_in[17];
    const float* n2_g    = (const float*)d_in[18];
    const float* n2_b    = (const float*)d_in[19];
    const float* bf1_w   = (const float*)d_in[20];
    const float* bf1_b   = (const float*)d_in[21];
    const float* bf2_w   = (const float*)d_in[22];
    const float* bf2_b   = (const float*)d_in[23];
    const float* ml_g    = (const float*)d_in[24];
    const float* ml_b    = (const float*)d_in[25];
    const float* f1_w    = (const float*)d_in[26];
    const float* f1_b    = (const float*)d_in[27];
    const float* f2_w    = (const float*)d_in[28];
    const float* f2_b    = (const float*)d_in[29];
    const float* fl_g    = (const float*)d_in[30];
    const float* fl_b    = (const float*)d_in[31];
    const float* fc_w    = (const float*)d_in[32];
    const float* fc_b    = (const float*)d_in[33];

    // ---- workspace bump allocator ----
    char* ws = (char*)d_ws;
    size_t off = 0;
    auto take = [&](size_t bytes) -> char* {
        char* p = ws + off;
        off += (bytes + 255) & ~(size_t)255;
        return p;
    };
    // persistent
    float*    hres  = (float*)   take((size_t)T * EMB * 4);
    _Float16* xn_h  = (_Float16*)take((size_t)T * EMB * 2);
    float*    msum  = (float*)   take((size_t)T * EMB * 4);
    _Float16* in_wT  = (_Float16*)take((size_t)1024 * 256 * 2);
    _Float16* xp_wT  = (_Float16*)take((size_t)48   * 512 * 2);
    _Float16* dt_wT  = (_Float16*)take((size_t)512  * 32  * 2);
    _Float16* out_wT = (_Float16*)take((size_t)256  * 512 * 2);
    _Float16* bf1_wT = (_Float16*)take((size_t)1024 * 256 * 2);
    _Float16* bf2_wT = (_Float16*)take((size_t)256  * 1024* 2);
    _Float16* f1_wT  = (_Float16*)take((size_t)1024 * 256 * 2);
    _Float16* f2_wT  = (_Float16*)take((size_t)256  * 1024* 2);
    _Float16* fc_wT  = (_Float16*)take((size_t)QPAD * 256 * 2);
    // mamba-phase scratch
    size_t scratch_base = off;
    float*    xz    = (float*)   take((size_t)T * 2*DIN * 4);
    float*    xs_f  = (float*)   take((size_t)T * DIN * 4);
    _Float16* xs_h  = (_Float16*)take((size_t)T * DIN * 2);
    float*    dbc   = (float*)   take((size_t)T * 48  * 4);
    _Float16* dtin  = (_Float16*)take((size_t)T * 32  * 2);
    float*    dtf   = (float*)   take((size_t)T * DIN * 4);
    _Float16* y_h   = (_Float16*)take((size_t)T * DIN * 2);
    // ffn-phase scratch (aliases mamba scratch; phases are sequential)
    off = scratch_base;
    _Float16* m_h   = (_Float16*)take((size_t)T * EMB * 2);
    _Float16* t1_h  = (_Float16*)take((size_t)T * 4*EMB * 2);
    float*    ffb   = (float*)   take((size_t)T * EMB * 4);
    float*    hid_f = (float*)   take((size_t)T * EMB * 4);
    _Float16* hid_h = (_Float16*)take((size_t)T * EMB * 2);
    float*    hs_f  = (float*)   take((size_t)T * EMB * 4);
    _Float16* hs2_h = (_Float16*)take((size_t)T * EMB * 2);

    auto tr = [&](const float* src, _Float16* dst, int K, int N, int Kpad, int Npad) {
        int total = Npad * Kpad;
        k_transpose_f16<<<dim3((total + 255) / 256), dim3(256), 0, stream>>>(src, dst, K, N, Kpad, Npad);
    };
    auto gemm = [&](const _Float16* A, const _Float16* BT, const float* bias,
                    float* Cf, _Float16* Ch, int M, int Npad, int K, int Nreal, int flags) {
        dim3 grid((Npad + 127) / 128, M / 32);
        k_gemm_wmma<<<grid, dim3(128), 0, stream>>>(A, BT, bias, Cf, Ch, M, Npad, K, Nreal, flags);
    };

    // ---- weights -> f16 transposed ----
    tr(in_w,  in_wT,  256,  1024, 256,  1024);
    tr(xp_w,  xp_wT,  512,  48,   512,  48);
    tr(dt_w,  dt_wT,  16,   512,  32,   512);   // K padded 16->32
    tr(out_w, out_wT, 512,  256,  512,  256);
    tr(bf1_w, bf1_wT, 256,  1024, 256,  1024);
    tr(bf2_w, bf2_wT, 1024, 256,  1024, 256);
    tr(f1_w,  f1_wT,  256,  1024, 256,  1024);
    tr(f2_w,  f2_wT,  1024, 256,  1024, 256);
    tr(fc_w,  fc_wT,  256,  QUES, 256,  QPAD);  // N padded 3162->3168

    // ---- h = LN0(qa_tab[qa]);  xn = LN1(h) ----
    k_embed_ln<<<dim3(T / 8), dim3(256), 0, stream>>>(qa, qa_tab, ln0_g, ln0_b, hres, T, 1e-12f);
    k_ln<<<dim3(T / 8), dim3(256), 0, stream>>>(hres, nullptr, n1_g, n1_b, nullptr, xn_h, T, 1e-5f);

    // ---- xz = xn @ in_w   (shared by both directions; token-pointwise) ----
    gemm(xn_h, in_wT, nullptr, xz, nullptr, T, 2*DIN, EMB, 2*DIN, 0);

    // ---- bidirectional mamba; msum accumulates fwd+bwd @ out_w ----
    for (int dir = 0; dir < 2; dir++) {
        k_conv_silu<<<dim3((BB*SS*DIN) / 256), dim3(256), 0, stream>>>(xz, conv_w, conv_b, xs_f, xs_h, dir);
        gemm(xs_h, xp_wT, nullptr, dbc, nullptr, T, 48, DIN, 48, 0);
        k_dtin<<<dim3((T * 32) / 256), dim3(256), 0, stream>>>(dbc, dtin, T);
        gemm(dtin, dt_wT, dt_b, dtf, nullptr, T, DIN, 32, DIN, GFLAG_SP);          // softplus fused
        k_scan<<<dim3(64), dim3(256), 0, stream>>>(dtf, xs_f, dbc, A_log, Dp, xz, y_h, dir);
        gemm(y_h, out_wT, nullptr, msum, nullptr, T, EMB, DIN, EMB, dir ? GFLAG_ACC : 0);
    }

    // ---- m = LN2(fwd+bwd); ff = gelu(m@bf1+b)@bf2+b; out = ff + h ----
    k_ln<<<dim3(T / 8), dim3(256), 0, stream>>>(msum, nullptr, n2_g, n2_b, nullptr, m_h, T, 1e-5f);
    gemm(m_h, bf1_wT, bf1_b, nullptr, t1_h, T, 4*EMB, EMB, 4*EMB, GFLAG_GELU);
    gemm(t1_h, bf2_wT, bf2_b, ffb, nullptr, T, EMB, 4*EMB, EMB, 0);

    // ---- hid = LN(ff + h); hs = gelu(hid@f1+b)@f2+b; hs2 = LN(hs + hid) ----
    k_ln<<<dim3(T / 8), dim3(256), 0, stream>>>(ffb, hres, ml_g, ml_b, hid_f, hid_h, T, 1e-12f);
    gemm(hid_h, f1_wT, f1_b, nullptr, t1_h, T, 4*EMB, EMB, 4*EMB, GFLAG_GELU);
    gemm(t1_h, f2_wT, f2_b, hs_f, nullptr, T, EMB, 4*EMB, EMB, 0);
    k_ln<<<dim3(T / 8), dim3(256), 0, stream>>>(hs_f, hid_f, fl_g, fl_b, nullptr, hs2_h, T, 1e-12f);

    // ---- logits = hs2 @ fc_w + fc_b  ->  d_out (T x 3162 f32) ----
    gemm(hs2_h, fc_wT, fc_b, (float*)d_out, nullptr, T, QPAD, EMB, QUES, 0);
}
